// Quantize_8564164788701
// MI455X (gfx1250) — matrix operand order repaired and checked by reading, not compiled, not run on hardware
//
#include <hip/hip_runtime.h>
#include <hip/hip_bf16.h>

// ---------------------------------------------------------------------------
// VQ nearest-code search for MI455X (gfx1250, wave32, WMMA).
//   dist(n,e) = ||x||^2 - 2 x.e + ||e||^2 ; argmin over e
//   -> maximize score = 2*(x.e) - ||e||^2   (||x||^2 constant per token)
// GEMM as split-precision bf16x3: x.e ~= xh.eh + xh.el + xl.eh
// with v_wmma_f32_16x16x32_bf16. A tile (16 tokens x 512 K, hi+lo) lives
// entirely in VGPRs; B streams through a 4-deep LDS ring filled with
// global_load_async_to_lds_b128 (ASYNCcnt path, no VGPR round-trip).
// Block = 8 waves = 4 M-tiles x 2 code-splits; 256 blocks = 2048 waves.
// ---------------------------------------------------------------------------

#define DIM     512
#define NEMB    8192
#define NTOK    16384           // 8 * 2048
#define NELEM   (NTOK * DIM)    // 8388608
#define EELEM   (NEMB * DIM)    // 4194304
#define NCHUNK  (NEMB / 16)     // 512 chunks of 16 codes
#define KSTEPS  (DIM / 32)      // 16 k-steps of 32

typedef __attribute__((ext_vector_type(16))) __bf16 v16bf;
typedef __attribute__((ext_vector_type(8)))  float  v8f;

union FragBF { uint4 u[2]; v16bf v; };

// ---------------- workspace layout (bytes) ----------------
#define OFF_NORM  ((size_t)0)                        // 8192 f32   = 32 KB
#define OFF_IDX   ((size_t)32768)                    // 16384 i32  = 64 KB
#define OFF_ACC   ((size_t)98304)                    // 1 f32 (+pad)
#define OFF_AHI   ((size_t)131072)                   // NTOK*DIM bf16 = 16 MB
#define OFF_ALO   (OFF_AHI + (size_t)NELEM * 2)      // 16 MB
#define OFF_BHI   (OFF_ALO + (size_t)NELEM * 2)      // NEMB*DIM bf16 = 8 MB
#define OFF_BLO   (OFF_BHI + (size_t)EELEM * 2)      // 8 MB
#define OFF_ET    (OFF_BLO + (size_t)EELEM * 2)      // embed.T f32 = 16 MB
// total ~64.2 MB

// ---------------- tiny init: zero the diff accumulator ----------------
__global__ void vq_init(float* acc) {
    if (threadIdx.x == 0 && blockIdx.x == 0) *acc = 0.0f;
}

// ---------------- per-code squared norms (coalesced over codes) ----------------
__global__ void vq_norms(const float* __restrict__ embed, float* __restrict__ norms) {
    int e = blockIdx.x * blockDim.x + threadIdx.x;
    if (e >= NEMB) return;
    float s = 0.0f;
    for (int k = 0; k < DIM; ++k) {
        float v = embed[(size_t)k * NEMB + e];
        s += v * v;
    }
    norms[e] = s;
}

// ---------------- embed: transpose to [code][k], split hi/lo bf16, fp32 copy ----
__global__ void vq_prep_embed(const float* __restrict__ embed,
                              __bf16* __restrict__ bHi, __bf16* __restrict__ bLo,
                              float* __restrict__ embedT) {
    int o = blockIdx.x * blockDim.x + threadIdx.x;
    if (o >= EELEM) return;
    int e = o >> 9;            // code
    int k = o & (DIM - 1);     // dim
    float v = embed[(size_t)k * NEMB + e];
    __bf16 h = (__bf16)v;
    bHi[o] = h;
    bLo[o] = (__bf16)(v - (float)h);
    embedT[o] = v;
}

// ---------------- input: split hi/lo bf16 (row-major [token][k]) ----------------
__global__ void vq_prep_input(const float* __restrict__ in,
                              __bf16* __restrict__ aHi, __bf16* __restrict__ aLo) {
    int i = blockIdx.x * blockDim.x + threadIdx.x;
    if (i >= NELEM) return;
    float x = in[i];
    __bf16 h = (__bf16)x;
    aHi[i] = h;
    aLo[i] = (__bf16)(x - (float)h);
}

// ---------------------------------------------------------------------------
// Fused GEMM + argmax.
// 256 threads = 8 waves: wave w -> M-tile (w&3), code-split (w>>2).
// Split s processes chunks c = s, s+2, s+4, ... (256 chunks each).
// LDS: 4 x 32 KB B ring (chunk c -> slot c&3; each slot = {hi 16K, lo 16K},
//      16 rows x 1024B, XOR-swizzled 16B granules) + 1 KB argmax scratch.
// B staged with global_load_async_to_lds_b128; one s_wait_asynccnt 0 per
// chunk pair, issued a full 48-WMMA compute pass ahead of consumption.
// A fragments (16 k-steps x hi/lo) are register-resident, loaded once.
// ---------------------------------------------------------------------------
#define LDS_BBUF  32768
#define LDS_BLO   16384
#define LDS_RED   131072
#define LDS_TOTAL (131072 + 64 * 2 * 8)

__global__ __launch_bounds__(256)
void vq_argmax(const __bf16* __restrict__ aHiG, const __bf16* __restrict__ aLoG,
               const __bf16* __restrict__ bHiG, const __bf16* __restrict__ bLoG,
               const float* __restrict__ norms, int* __restrict__ bestIdx) {
    extern __shared__ char smem[];
    char*  bL   = smem;
    float* redV = (float*)(smem + LDS_RED);
    int*   redI = (int*)  (smem + LDS_RED + 64 * 2 * 4);

    const int tid   = threadIdx.x;
    const int lane  = tid & 31;
    const int wave  = tid >> 5;
    const int mtile = wave & 3;
    const int split = wave >> 2;   // 0: even chunks, 1: odd chunks
    const int h     = lane >> 4;   // K-half selector
    const int l16   = lane & 15;   // A row / B code / N column
    const int tokBase = blockIdx.x * 64;

    // ---- load this wave's A tile into registers: 16 k-steps x (hi,lo) ----
    // Lane holds row l16; K in [8h,8h+8) u [16+8h,16+8h+8) per 32-wide step.
    FragBF aH[KSTEPS], aL[KSTEPS];
    {
        const size_t rowOff = (size_t)(tokBase + mtile * 16 + l16) * 1024 + (size_t)(h * 16);
        const char* aHiRow = (const char*)aHiG + rowOff;
        const char* aLoRow = (const char*)aLoG + rowOff;
        #pragma unroll
        for (int ks = 0; ks < KSTEPS; ++ks) {
            aH[ks].u[0] = *(const uint4*)(aHiRow + ks * 64);
            aH[ks].u[1] = *(const uint4*)(aHiRow + ks * 64 + 32);
            aL[ks].u[0] = *(const uint4*)(aLoRow + ks * 64);
            aL[ks].u[1] = *(const uint4*)(aLoRow + ks * 64 + 32);
        }
    }

    // LDS base offset of the B ring (low 32 bits of generic LDS address)
    const uint32_t bBase = (uint32_t)(size_t)bL;

    // ---- async B stager: 16 codes * 512 K * (hi+lo) = 2048 x 16B granules ----
    // Direct memory -> LDS DMA per lane; tracked by ASYNCcnt, no VGPRs touched.
    auto stageB = [&](int c) {
        const uint32_t slot = bBase + (uint32_t)((c & 3) * LDS_BBUF);
        #pragma unroll
        for (int i = 0; i < 8; ++i) {
            int g   = tid + i * 256;       // 0..2047
            int arr = g >> 10;             // 0 = hi, 1 = lo
            int idx = g & 1023;
            int row = idx >> 6;            // code within chunk
            int gr  = idx & 63;
            const char* src = (const char*)(arr ? bLoG : bHiG)
                              + (size_t)(c * 16 + row) * 1024 + (size_t)gr * 16;
            uint32_t dst = slot + (uint32_t)(arr * LDS_BLO + row * 1024 + ((gr ^ row) << 4));
            asm volatile("global_load_async_to_lds_b128 %0, %1, off"
                         :: "v"(dst), "v"(src) : "memory");
        }
    };

    float bestv[8];
    int   besti[8];
    #pragma unroll
    for (int r = 0; r < 8; ++r) { bestv[r] = -3.0e38f; besti[r] = 0; }

    stageB(0);
    stageB(1);
    asm volatile("s_wait_asynccnt 0x0" ::: "memory");
    __syncthreads();

    const int sw = l16;                    // LDS XOR-swizzle key

    for (int j = 0; j < NCHUNK / 2; ++j) {
        if (j + 1 < NCHUNK / 2) {          // async prefetch of next chunk pair
            stageB(2 * j + 2);
            stageB(2 * j + 3);
        }

        const int c = 2 * j + split;       // this wave's chunk
        const char* bHiRow = bL + (c & 3) * LDS_BBUF + l16 * 1024;

        v8f acc = {0.f, 0.f, 0.f, 0.f, 0.f, 0.f, 0.f, 0.f};
        #pragma unroll
        for (int ks = 0; ks < KSTEPS; ++ks) {
            FragBF bh, blo;
            // B 32x16 fragment: lane holds col l16, K in [8h,8h+8) u [16+8h,16+8h+8)
            const int gb0 = ks * 4 + h;
            const int o0  = ((gb0       ^ sw) << 4);
            const int o1  = (((gb0 + 2) ^ sw) << 4);
            bh.u[0]  = *(const uint4*)(bHiRow + o0);
            bh.u[1]  = *(const uint4*)(bHiRow + o1);
            blo.u[0] = *(const uint4*)(bHiRow + LDS_BLO + o0);
            blo.u[1] = *(const uint4*)(bHiRow + LDS_BLO + o1);
            // x.e ~= xh.eh + xh.el + xl.eh  (bf16x3 split-precision)
            acc = __builtin_amdgcn_wmma_f32_16x16x32_bf16(false, aH[ks].v, false, bh.v,
                                                          (short)0, acc, false, false);
            acc = __builtin_amdgcn_wmma_f32_16x16x32_bf16(false, aH[ks].v, false, blo.v,
                                                          (short)0, acc, false, false);
            acc = __builtin_amdgcn_wmma_f32_16x16x32_bf16(false, aL[ks].v, false, bh.v,
                                                          (short)0, acc, false, false);
        }

        const int   code = c * 16 + l16;
        const float nrm  = norms[code];
        #pragma unroll
        for (int r = 0; r < 8; ++r) {
            float s = 2.0f * acc[r] - nrm;          // = -dist + const(token)
            if (s > bestv[r]) { bestv[r] = s; besti[r] = code; }
        }

        // staged pair for j+1 must be complete before the publishing barrier
        asm volatile("s_wait_asynccnt 0x0" ::: "memory");
        __syncthreads();
    }

    // ---- argmax merge: 16 N-lanes via shuffles, then across the 2 splits ----
    #pragma unroll
    for (int r = 0; r < 8; ++r) {
        float v = bestv[r];
        int   i = besti[r];
        #pragma unroll
        for (int off = 1; off < 16; off <<= 1) {
            float ov = __shfl_xor(v, off, 32);
            int   oi = __shfl_xor(i, off, 32);
            if (ov > v || (ov == v && oi < i)) { v = ov; i = oi; }
        }
        if (l16 == 0) {
            int t = mtile * 16 + r + 8 * h;         // token within block
            redV[t * 2 + split] = v;
            redI[t * 2 + split] = i;
        }
    }
    __syncthreads();
    if (tid < 64) {
        float v0 = redV[tid * 2], v1 = redV[tid * 2 + 1];
        int   i0 = redI[tid * 2], i1 = redI[tid * 2 + 1];
        // ties -> lowest index (matches jnp.argmax first-max semantics)
        int win = (v1 > v0 || (v1 == v0 && i1 < i0)) ? i1 : i0;
        bestIdx[tokBase + tid] = win;
    }
}

// ---------------- gather quantize + MSE partial reduction ----------------
__global__ __launch_bounds__(256)
void vq_gather(const float* __restrict__ in, const float* __restrict__ embedT,
               const int* __restrict__ bestIdx, float* __restrict__ out,
               float* __restrict__ diffAcc) {
    __shared__ float red[256];
    float local = 0.0f;
    for (int e = blockIdx.x * 256 + threadIdx.x; e < NELEM; e += gridDim.x * 256) {
        int t = e >> 9;
        int d = e & (DIM - 1);
        float q = embedT[(size_t)bestIdx[t] * DIM + d];
        float x = in[e];
        out[e] = q;                       // quantize_st == quantize numerically
        float df = q - x;
        local += df * df;
    }
    red[threadIdx.x] = local;
    __syncthreads();
    for (int s = 128; s > 0; s >>= 1) {
        if (threadIdx.x < s) red[threadIdx.x] += red[threadIdx.x + s];
        __syncthreads();
    }
    if (threadIdx.x == 0) atomicAdd(diffAcc, red[0]);
}

// ---------------- finalize: diff scalar + indices-as-float ----------------
__global__ void vq_finalize(const int* __restrict__ bestIdx,
                            const float* __restrict__ diffAcc,
                            float* __restrict__ out) {
    int t = blockIdx.x * blockDim.x + threadIdx.x;
    if (t < NTOK) out[NELEM + 1 + t] = (float)bestIdx[t];
    if (t == 0)   out[NELEM] = diffAcc[0] * (1.0f / (float)NELEM);
}

// ---------------------------------------------------------------------------
extern "C" void kernel_launch(void* const* d_in, const int* in_sizes, int n_in,
                              void* d_out, int out_size, void* d_ws, size_t ws_size,
                              hipStream_t stream) {
    const float* input = (const float*)d_in[0];   // [8,2048,512] fp32
    const float* embed = (const float*)d_in[1];   // [512,8192]  fp32
    float* out = (float*)d_out;                   // [q(8.4M) | diff | ind(16K)]

    char* ws = (char*)d_ws;
    float*  norms   = (float*)(ws + OFF_NORM);
    int*    bestIdx = (int*)  (ws + OFF_IDX);
    float*  diffAcc = (float*)(ws + OFF_ACC);
    __bf16* aHi     = (__bf16*)(ws + OFF_AHI);
    __bf16* aLo     = (__bf16*)(ws + OFF_ALO);
    __bf16* bHi     = (__bf16*)(ws + OFF_BHI);
    __bf16* bLo     = (__bf16*)(ws + OFF_BLO);
    float*  embedT  = (float*)(ws + OFF_ET);

    vq_init<<<1, 32, 0, stream>>>(diffAcc);
    vq_norms<<<NEMB / 256, 256, 0, stream>>>(embed, norms);
    vq_prep_embed<<<EELEM / 256, 256, 0, stream>>>(embed, bHi, bLo, embedT);
    vq_prep_input<<<NELEM / 256, 256, 0, stream>>>(input, aHi, aLo);

    // 256 blocks x 8 waves; ~129 KB dynamic LDS (B ring + argmax scratch)
    vq_argmax<<<NTOK / 64, 256, LDS_TOTAL, stream>>>(aHi, aLo, bHi, bLo,
                                                     norms, bestIdx);

    vq_gather<<<2048, 256, 0, stream>>>(input, embedT, bestIdx, out, diffAcc);
    vq_finalize<<<(NTOK + 255) / 256, 256, 0, stream>>>(bestIdx, diffAcc, out);
}
